// UltraFastShiftNet_32512902431269
// MI455X (gfx1250) — matrix operand order repaired and checked by reading, not compiled, not run on hardware
//
#include <hip/hip_runtime.h>
#include <hip/hip_bf16.h>
#include <math.h>

typedef __attribute__((ext_vector_type(16))) _Float16 v16h;
typedef __attribute__((ext_vector_type(8)))  _Float16 v8h;
typedef __attribute__((ext_vector_type(8)))  float    v8f;
typedef __attribute__((ext_vector_type(4)))  float    v4f;

union AV { v16h v; v8h h[2]; };

#define XLD 528   // f16 activation LDS row stride (rows stay 16B aligned)
#define YLD 520   // f32 GEMM-output LDS row stride (rows stay 16B aligned)
#define PLD 68    // 64-wide f32 tiles stride

// LDS carve offsets (bytes)
#define OFF_X   0
#define OFF_Y   (OFF_X + 64*XLD*2)        // 67584
#define OFF_A   (OFF_Y + 64*YLD*4)        // 200704
#define OFF_P   (OFF_A + 64*PLD*4)        // 218112
#define OFF_O   (OFF_P + 64*PLD*4)        // 235520
#define OFF_R   (OFF_O + 64*PLD*4)        // 252928
#define SMEM_BYTES (OFF_R + 256*4)        // 253952 < 320KB WGP LDS

// ---------------------------------------------------------------------------
// One-time repack: f32 row-major W[K][N] -> f16 WMMA B-fragment-major layout.
// Fragment f = kt*(N/16)+nt holds 32 lanes x 16 contiguous f16 each:
//   lane l: n = nt*16 + (l&15), elem e -> k = kt*32+(e>>3)*16+(l>>4)*8+(e&7)
// ---------------------------------------------------------------------------
__global__ void repack_weights(const float* __restrict__ src,
                               _Float16* __restrict__ dst, int K, int N) {
  int o = blockIdx.x * 256 + threadIdx.x;
  if (o >= K * N) return;
  int nnT  = N >> 4;
  int f    = o >> 9;           // 512 f16 per fragment
  int lane = (o >> 4) & 31;
  int e    = o & 15;
  int kt   = f / nnT;
  int nt   = f - kt * nnT;
  int k = kt * 32 + ((e >> 3) << 4) + ((lane >> 4) << 3) + (e & 7);
  int n = (nt << 4) + (lane & 15);
  dst[o] = (_Float16)src[k * N + n];
}

// ---------------------------------------------------------------------------
// A-fragment load (ISA 16-bit A-layout): lane m=l&15; e<8 -> k=kb+(l>=16?8:0)+e,
// e>=8 -> +16.  Two ds_load_b128 per fragment.
// ---------------------------------------------------------------------------
__device__ __forceinline__ AV load_afrag(const _Float16* __restrict__ sA,
                                         int m, int kb, int l15, int lh) {
  AV a;
  const _Float16* p = sA + (m + l15) * XLD + kb + lh * 8;
  a.h[0] = *(const v8h*)(p);
  a.h[1] = *(const v8h*)(p + 16);
  return a;
}

// ---------------------------------------------------------------------------
// M-split GEMM: wave owns rows [16*wave, 16*wave+16), loops all NTILES n-tiles.
// TWO independent accumulator chains (even/odd k-chunks) hide the
// WMMA->WMMA RAW hazard window; chains are combined at the C-store together
// with the (optional) bias, so no extra VALU vs. a single chain.
// ---------------------------------------------------------------------------
template<int KCH, int NTILES, bool BIAS>
__device__ __forceinline__ void gemm_mtile(const _Float16* __restrict__ sA,
                                           const _Float16* __restrict__ Bfrag,
                                           const float* __restrict__ bias,
                                           float* __restrict__ sC,
                                           int wave, int lane) {
  static_assert((KCH & 1) == 0, "KCH must be even");
  const int m0  = wave << 4;
  const int l15 = lane & 15;
  const int lh  = lane >> 4;

  AV afr[KCH];
#pragma unroll
  for (int kc = 0; kc < KCH; ++kc)
    afr[kc] = load_afrag(sA, m0, kc * 32, l15, lh);

  const v16h* Bv = (const v16h*)Bfrag;
#pragma unroll 2
  for (int nt = 0; nt < NTILES; ++nt) {
    if (nt + 1 < NTILES)
      __builtin_prefetch((const void*)(Bv + ((nt + 1) * 32 + lane)), 0, 3);

    v8f acc0 = {0.f, 0.f, 0.f, 0.f, 0.f, 0.f, 0.f, 0.f};
    v8f acc1 = {0.f, 0.f, 0.f, 0.f, 0.f, 0.f, 0.f, 0.f};
#pragma unroll
    for (int kc = 0; kc < KCH; kc += 2) {
      v16h bf0 = Bv[(((kc + 0) * NTILES + nt) * 32 + lane)];
      v16h bf1 = Bv[(((kc + 1) * NTILES + nt) * 32 + lane)];
      acc0 = __builtin_amdgcn_wmma_f32_16x16x32_f16(
          false, afr[kc + 0].v, false, bf0, (short)0, acc0, false, false);
      acc1 = __builtin_amdgcn_wmma_f32_16x16x32_f16(
          false, afr[kc + 1].v, false, bf1, (short)0, acc1, false, false);
    }
    const int n = (nt << 4) + l15;
    const float bv = BIAS ? bias[n] : 0.0f;
#pragma unroll
    for (int v = 0; v < 8; ++v) {
      float r = acc0[v] + acc1[v];
      if (BIAS) r += bv;
      sC[(m0 + v + 8 * lh) * YLD + n] = r;
    }
  }
}

// ---------------------------------------------------------------------------
// K-split GEMM: wave owns k-chunks [wave*KCW, wave*KCW+KCW) for ALL four
// 16-row m-tiles -> each B fragment is loaded once per block and reused 4x
// from registers (4x less L2 weight traffic).  The four per-m chains are
// independent (unrolled), covering the WMMA hazard window.  Partials are
// combined with ds_add_f32 LDS atomics into zero-initialized sC.
// ---------------------------------------------------------------------------
template<int KCW, int NTILES>
__device__ __forceinline__ void gemm_ksplit(const _Float16* __restrict__ sA,
                                            const _Float16* __restrict__ Bfrag,
                                            float* __restrict__ sC,
                                            int wave, int lane) {
  const int l15 = lane & 15;
  const int lh  = lane >> 4;
  const int kc0 = wave * KCW;

  AV afr[4][KCW];
#pragma unroll
  for (int m = 0; m < 4; ++m)
#pragma unroll
    for (int kc = 0; kc < KCW; ++kc)
      afr[m][kc] = load_afrag(sA, m << 4, (kc0 + kc) * 32, l15, lh);

  const v16h* Bv = (const v16h*)Bfrag;
  for (int nt = 0; nt < NTILES; ++nt) {
    if (nt + 1 < NTILES)
      __builtin_prefetch((const void*)(Bv + ((kc0 * NTILES + nt + 1) * 32 + lane)), 0, 3);

    v16h bf[KCW];
#pragma unroll
    for (int kc = 0; kc < KCW; ++kc)
      bf[kc] = Bv[(((kc0 + kc) * NTILES + nt) * 32 + lane)];

    const int n = (nt << 4) + l15;
#pragma unroll
    for (int m = 0; m < 4; ++m) {
      v8f acc = {0.f, 0.f, 0.f, 0.f, 0.f, 0.f, 0.f, 0.f};
#pragma unroll
      for (int kc = 0; kc < KCW; ++kc)
        acc = __builtin_amdgcn_wmma_f32_16x16x32_f16(
            false, afr[m][kc].v, false, bf[kc], (short)0, acc, false, false);
#pragma unroll
      for (int v = 0; v < 8; ++v)
        (void)__hip_atomic_fetch_add(&sC[((m << 4) + v + 8 * lh) * YLD + n],
                                     acc[v], __ATOMIC_RELAXED,
                                     __HIP_MEMORY_SCOPE_WORKGROUP);
    }
  }
}

__device__ __forceinline__ float gelu_exact(float x) {
  return 0.5f * x * (1.0f + erff(x * 0.70710678118f));
}

// Zero the 64x512 live region of the f32 scratch (float4 stores).
__device__ __forceinline__ void zero_sy(float* __restrict__ sY, int tid) {
  for (int idx = tid; idx < 64 * 128; idx += 128) {
    int r = idx >> 7, c4 = (idx & 127) << 2;
    *(v4f*)(sY + r * YLD + c4) = (v4f){0.f, 0.f, 0.f, 0.f};
  }
}

// bias-add + LayerNorm(512) + exact gelu, f32 LDS -> f16 LDS. 2 threads/row.
__device__ void ln_gelu_512(const float* __restrict__ sY,
                            _Float16* __restrict__ sX,
                            const float* __restrict__ bias,
                            const float* __restrict__ g,
                            const float* __restrict__ bt,
                            float* __restrict__ red, int tid) {
  const int row = tid >> 1, h = tid & 1;
  const float* y  = sY + row * YLD + h * 256;
  const float* bb = bias + h * 256;
  float s = 0.f, ss = 0.f;
  for (int c = 0; c < 256; ++c) { float v = y[c] + bb[c]; s += v; ss += v * v; }
  red[tid] = s; red[128 + tid] = ss;
  __syncthreads();
  float S  = red[tid] + red[tid ^ 1];
  float SS = red[128 + tid] + red[128 + (tid ^ 1)];
  float mu  = S * (1.0f / 512.0f);
  float var = SS * (1.0f / 512.0f) - mu * mu;
  float rs  = rsqrtf(var + 1e-5f);
  _Float16* x = sX + row * XLD + h * 256;
  const float* gg = g + h * 256;
  const float* tt = bt + h * 256;
  for (int c = 0; c < 256; ++c)
    x[c] = (_Float16)gelu_exact((y[c] + bb[c] - mu) * rs * gg[c] + tt[c]);
  __syncthreads();
}

// ---------------------------------------------------------------------------
// Fully fused pipeline: 1 block = 128 threads = 4 waves = 64 rows.
// ---------------------------------------------------------------------------
__global__ void __launch_bounds__(128) shiftnet_fused(
    const float* __restrict__ a_bits, const float* __restrict__ shift_bits,
    const _Float16* __restrict__ W1f, const float* __restrict__ b1,
    const float* __restrict__ g1, const float* __restrict__ bt1,
    const _Float16* __restrict__ W2f, const float* __restrict__ b2,
    const float* __restrict__ g2, const float* __restrict__ bt2,
    const _Float16* __restrict__ W3f, const float* __restrict__ b3,
    const _Float16* __restrict__ Wr1f, const float* __restrict__ br1,
    const _Float16* __restrict__ Wr2f, const float* __restrict__ br2,
    const float* __restrict__ temp_p, float* __restrict__ out) {
  extern __shared__ char smem[];
  _Float16* s_x = (_Float16*)(smem + OFF_X);  // activations (f16), 64 x <=512
  float*    s_y = (float*)(smem + OFF_Y);     // GEMM out (f32),    64 x 512
  float*    s_a = (float*)(smem + OFF_A);     // a_bits tile        64 x 64
  float*    s_p = (float*)(smem + OFF_P);     // shift_probs        64 x 64
  float*    s_o = (float*)(smem + OFF_O);     // pre-correction out 64 x 64
  float*    s_r = (float*)(smem + OFF_R);     // LN reductions

  const int tid  = threadIdx.x;
  const int wave = tid >> 5;
  const int lane = tid & 31;
  const long row0 = (long)blockIdx.x * 64;

  // ---- stage 0: stage inputs into LDS ----
  for (int idx = tid; idx < 64 * 64; idx += 128) {
    int r = idx >> 6, c = idx & 63;
    s_x[r * XLD + c] = (_Float16)shift_bits[(row0 + r) * 64 + c];
    s_a[r * PLD + c] = a_bits[(row0 + r) * 64 + c];
  }
  __syncthreads();

  // ---- h = gelu(LN(x @ W1 + b1)) : 64x64 @ 64x512 (bias folded into LN) ----
  gemm_mtile<2, 32, false>(s_x, W1f, nullptr, s_y, wave, lane);
  __syncthreads();
  ln_gelu_512(s_y, s_x, b1, g1, bt1, s_r, tid);

  // ---- h = gelu(LN(h @ W2 + b2)) : 64x512 @ 512x512, K-split 4 ways ----
  zero_sy(s_y, tid);
  __syncthreads();
  gemm_ksplit<4, 32>(s_x, W2f, s_y, wave, lane);
  __syncthreads();
  ln_gelu_512(s_y, s_x, b2, g2, bt2, s_r, tid);

  // ---- logits = h @ W3 + b3 : 64x512 @ 512x64 ----
  gemm_mtile<16, 4, true>(s_x, W3f, b3, s_y, wave, lane);
  __syncthreads();

  // ---- softmax(logits / T) -> s_p (f32) and s_x[:,64:128] (f16) ----
  const float invT = 1.0f / temp_p[0];
  if (tid < 64) {
    const int r = tid;
    float mx = -3.4e38f;
    for (int c = 0; c < 64; ++c) mx = fmaxf(mx, s_y[r * YLD + c] * invT);
    float sum = 0.f;
    for (int c = 0; c < 64; ++c) {
      float e = __expf(s_y[r * YLD + c] * invT - mx);
      s_p[r * PLD + c] = e; sum += e;
    }
    float inv = 1.0f / sum;
    for (int c = 0; c < 64; ++c) {
      float p = s_p[r * PLD + c] * inv;
      s_p[r * PLD + c] = p;
      s_x[r * XLD + 64 + c] = (_Float16)p;  // second half of r = concat(out, probs)
    }
  }
  __syncthreads();

  // ---- shift einsum collapses to causal correlation:
  //      out[b,i] = sum_{s<=i} probs[b,s] * a[b,i-s]  (M[s,i,j]=(i-j==s)) ----
  for (int idx = tid; idx < 64 * 64; idx += 128) {
    int r = idx >> 6, i = idx & 63;
    const float* pr = s_p + r * PLD;
    const float* ar = s_a + r * PLD;
    float acc = 0.f;
    for (int s = 0; s <= i; ++s) acc += pr[s] * ar[i - s];
    s_o[r * PLD + i] = acc;
    s_x[r * XLD + i] = (_Float16)acc;       // first half of r
  }
  __syncthreads();

  // ---- gelu(r @ Wr1 + br1) : 64x128 @ 128x512, K-split (1 chunk/wave) ----
  zero_sy(s_y, tid);
  __syncthreads();
  gemm_ksplit<1, 32>(s_x, Wr1f, s_y, wave, lane);
  __syncthreads();
  for (int idx = tid; idx < 64 * 512; idx += 128) {
    int r = idx >> 9, c = idx & 511;
    s_x[r * XLD + c] = (_Float16)gelu_exact(s_y[r * YLD + c] + br1[c]);
  }
  __syncthreads();

  // ---- correction logits : 64x512 @ 512x64 ----
  gemm_mtile<16, 4, true>(s_x, Wr2f, br2, s_y, wave, lane);
  __syncthreads();

  // ---- sigmoid + blend + clip + store ----
  for (int idx = tid; idx < 64 * 64; idx += 128) {
    int r = idx >> 6, c = idx & 63;
    float corr = 1.0f / (1.0f + __expf(-s_y[r * YLD + c]));
    float v = s_o[r * PLD + c] + 0.1f * (corr - 0.5f);
    v = fminf(fmaxf(v, 0.0f), 1.0f);
    out[(row0 + r) * 64 + c] = v;
  }
}

// ---------------------------------------------------------------------------
extern "C" void kernel_launch(void* const* d_in, const int* in_sizes, int n_in,
                              void* d_out, int out_size, void* d_ws, size_t ws_size,
                              hipStream_t stream) {
  const float* a_bits     = (const float*)d_in[0];
  const float* shift_bits = (const float*)d_in[1];
  const float* W1  = (const float*)d_in[2];
  const float* b1  = (const float*)d_in[3];
  const float* g1  = (const float*)d_in[4];
  const float* bt1 = (const float*)d_in[5];
  const float* W2  = (const float*)d_in[6];
  const float* b2  = (const float*)d_in[7];
  const float* g2  = (const float*)d_in[8];
  const float* bt2 = (const float*)d_in[9];
  const float* W3  = (const float*)d_in[10];
  const float* b3  = (const float*)d_in[11];
  const float* Wr1 = (const float*)d_in[12];
  const float* br1 = (const float*)d_in[13];
  const float* Wr2 = (const float*)d_in[14];
  const float* br2 = (const float*)d_in[15];
  const float* temp = (const float*)d_in[17];
  float* out = (float*)d_out;

  // f16 fragment-packed weights in workspace (rewritten every call; <1 MB)
  char* ws = (char*)d_ws;
  _Float16* W1f  = (_Float16*)(ws + 0);        // 64*512
  _Float16* W2f  = (_Float16*)(ws + 65536);    // 512*512
  _Float16* W3f  = (_Float16*)(ws + 589824);   // 512*64
  _Float16* Wr1f = (_Float16*)(ws + 655360);   // 128*512
  _Float16* Wr2f = (_Float16*)(ws + 786432);   // 512*64

  repack_weights<<<(64 * 512 + 255) / 256, 256, 0, stream>>>(W1, W1f, 64, 512);
  repack_weights<<<(512 * 512 + 255) / 256, 256, 0, stream>>>(W2, W2f, 512, 512);
  repack_weights<<<(512 * 64 + 255) / 256, 256, 0, stream>>>(W3, W3f, 512, 64);
  repack_weights<<<(128 * 512 + 255) / 256, 256, 0, stream>>>(Wr1, Wr1f, 128, 512);
  repack_weights<<<(512 * 64 + 255) / 256, 256, 0, stream>>>(Wr2, Wr2f, 512, 64);

  (void)hipFuncSetAttribute((const void*)shiftnet_fused,
                            hipFuncAttributeMaxDynamicSharedMemorySize,
                            (int)SMEM_BYTES);

  const int Btotal = in_sizes[0] / 64;          // 65536
  const int nblocks = (Btotal + 63) / 64;       // 1024 blocks x 128 threads
  shiftnet_fused<<<nblocks, 128, SMEM_BYTES, stream>>>(
      a_bits, shift_bits, W1f, b1, g1, bt1, W2f, b2, g2, bt2,
      W3f, b3, Wr1f, br1, Wr2f, br2, temp, out);
}
